// KernelModulatedLatentAttention_53480932770342
// MI455X (gfx1250) — compile-verified, matrix operand order checked by reading, not run on hardware
//
#include <hip/hip_runtime.h>
#include <hip/hip_bf16.h>

typedef unsigned short u16t;
typedef __attribute__((ext_vector_type(16))) __bf16 bf16x16;
typedef __attribute__((ext_vector_type(8)))  float  f32x8;

#define STATE_DIM 1024
#define CODE_DIM  384
#define NUM_HEADS 16
#define HEAD_DIM  64
#define INNER_DIM 1024
#define SIG_DIM   64
#define BB 8
#define UU 512
#define VV 512

__device__ __forceinline__ u16t f2bf(float x) {
    union { float f; unsigned u; } c; c.f = x;
    unsigned b = c.u + 0x7FFFu + ((c.u >> 16) & 1u);
    return (u16t)(b >> 16);
}
__device__ __forceinline__ float bf2f(u16t h) {
    union { unsigned u; float f; } c; c.u = ((unsigned)h) << 16;
    return c.f;
}

union FragB {
    bf16x16 v;
    uint4 q[2];
    u16t u[16];
};

// Load a 16xK-step (K=32) bf16 fragment (A or B operand, NT layout, K contiguous).
// lanes 0-15: row=lane,    elems 0..7 = K(0..7),  8..15 = K(16..23)
// lanes 16-31: row=lane-16, elems 0..7 = K(8..15), 8..15 = K(24..31)
template<typename PT>
__device__ __forceinline__ bf16x16 load_frag_p(PT tile, int ld, int lane) {
    int m = lane & 15;
    int kofs = (lane >> 4) * 8;
    FragB f;
    f.q[0] = *(const uint4*)(tile + (long)m * ld + kofs);
    f.q[1] = *(const uint4*)(tile + (long)m * ld + kofs + 16);
    return f.v;
}
__device__ __forceinline__ bf16x16 load_frag(const u16t* tile, int ld, int lane) {
    return load_frag_p(tile, ld, lane);
}

__device__ __forceinline__ f32x8 wmma_bf16(bf16x16 a, bf16x16 b, f32x8 c) {
    return __builtin_amdgcn_wmma_f32_16x16x32_bf16(false, a, false, b, (short)0, c, false, false);
}

// CDNA5 async global->LDS copy of 16 bytes per lane (ASYNCcnt-tracked).
__device__ __forceinline__ void async_g2l_b128(unsigned lds_addr, const u16t* gaddr) {
    asm volatile("global_load_async_to_lds_b128 %0, %1, off"
                 :: "v"(lds_addr), "v"((unsigned long long)(uintptr_t)gaddr)
                 : "memory");
}
__device__ __forceinline__ void wait_async0() {
    asm volatile("s_wait_asynccnt 0" ::: "memory");
}

// ---------------- elementwise convert f32 -> bf16 ----------------
__global__ void cvt_bf16(const float* __restrict__ in, u16t* __restrict__ out, int n) {
    int i = blockIdx.x * blockDim.x + threadIdx.x;
    int stride = gridDim.x * blockDim.x;
    for (; i < n; i += stride) out[i] = f2bf(in[i]);
}

// ---------------- layernorm rows of 1024, write bf16 ----------------
__global__ __launch_bounds__(256)
void layernorm_k(const float* __restrict__ x, const float* __restrict__ g,
                 const float* __restrict__ b, u16t* __restrict__ out) {
    __shared__ float s1[256], s2[256];
    long row = blockIdx.x;
    const float* xr = x + row * STATE_DIM;
    int t = threadIdx.x;
    float v0 = xr[t], v1 = xr[t + 256], v2 = xr[t + 512], v3 = xr[t + 768];
    float s = v0 + v1 + v2 + v3;
    float sq = v0*v0 + v1*v1 + v2*v2 + v3*v3;
    s1[t] = s; s2[t] = sq;
    __syncthreads();
    for (int o = 128; o > 0; o >>= 1) {
        if (t < o) { s1[t] += s1[t + o]; s2[t] += s2[t + o]; }
        __syncthreads();
    }
    float mean = s1[0] * (1.0f / STATE_DIM);
    float var  = s2[0] * (1.0f / STATE_DIM) - mean * mean;
    float rstd = rsqrtf(var + 1e-5f);
    u16t* orow = out + row * STATE_DIM;
    orow[t]       = f2bf((v0 - mean) * rstd * g[t]       + b[t]);
    orow[t + 256] = f2bf((v1 - mean) * rstd * g[t + 256] + b[t + 256]);
    orow[t + 512] = f2bf((v2 - mean) * rstd * g[t + 512] + b[t + 512]);
    orow[t + 768] = f2bf((v3 - mean) * rstd * g[t + 768] + b[t + 768]);
}

// ---------------- row log-softmax over 512, in place f32 ----------------
__global__ __launch_bounds__(256)
void logsoftmax_k(float* __restrict__ x) {
    __shared__ float sm[256];
    long row = blockIdx.x;
    float* xr = x + row * (long)VV;
    int t = threadIdx.x;
    float a = xr[t], b2 = xr[t + 256];
    float m = fmaxf(a, b2);
    sm[t] = m; __syncthreads();
    for (int o = 128; o > 0; o >>= 1) { if (t < o) sm[t] = fmaxf(sm[t], sm[t + o]); __syncthreads(); }
    float rm = sm[0];
    __syncthreads();
    float s = __expf(a - rm) + __expf(b2 - rm);
    sm[t] = s; __syncthreads();
    for (int o = 128; o > 0; o >>= 1) { if (t < o) sm[t] += sm[t + o]; __syncthreads(); }
    float lse = rm + __logf(sm[0]);
    xr[t] = a - lse;
    xr[t + 256] = b2 - lse;
}

// ---------------- generic NT bf16 WMMA GEMM, 64x64 block tile ----------------
// B tile (64 x 32 k-step) is staged once per block into LDS via CDNA5 async
// global->LDS copies, double-buffered and overlapped with the WMMAs.
struct EpiArgs {
    const float* bias;
    const u16t*  mul;     // bf16 elementwise multiplicand for MOD epilogue
    u16t*        obf;     // bf16 output
    float*       of;      // f32 output
    const float* resid;
    const float* scale;
    float alpha;
    int   ldc;
    long  cstride;        // per-batch C stride (elements)
};

// EPI: 0=LOGITS(f32*alpha, batched)  1=MOD(bf16: mul*(1+acc))
//      2=BIAS(bf16: acc+bias[n])     3=VTRANS(bf16 transposed per-head)
//      4=RESID(f32: resid + scale[n]*(acc+bias[n]))
template<int EPI>
__global__ __launch_bounds__(128)
void gemm_nt(const u16t* __restrict__ A, const u16t* __restrict__ Bw,
             int K, int lda, int ldb, long aStride, long bStride, EpiArgs e) {
    __shared__ u16t bs[2][64 * 32];     // double-buffered B tile, 2 x 4KB
    int wave = threadIdx.x >> 5;
    int lane = threadIdx.x & 31;
    int bz = blockIdx.z;
    const u16t* Ab = A  + (long)bz * aStride;
    const u16t* Bb = Bw + (long)bz * bStride;
    int row0 = blockIdx.y * 64 + wave * 16;
    int col0 = blockIdx.x * 64;

    // async-stage one 64x32 B tile (4KB): 256 chunks of 16B, 2 per thread
    auto issue = [&](int buf, int k) {
        #pragma unroll
        for (int ii = 0; ii < 2; ii++) {
            int ch = threadIdx.x + ii * 128;
            int r  = ch >> 2;              // B row (output column) 0..63
            int co = (ch & 3) * 8;         // k sub-offset in elements
            const u16t* g = Bb + (long)(col0 + r) * ldb + k + co;
            unsigned lds = (unsigned)(uintptr_t)&bs[buf][r * 32 + co];
            async_g2l_b128(lds, g);
        }
    };

    f32x8 acc[4];
    #pragma unroll
    for (int i = 0; i < 4; i++)
        #pragma unroll
        for (int j = 0; j < 8; j++) acc[i][j] = 0.0f;

    issue(0, 0);
    int buf = 0;
    for (int k = 0; k < K; k += 32) {
        wait_async0();
        __syncthreads();
        if (k + 32 < K) issue(buf ^ 1, k + 32);
        bf16x16 a = load_frag(Ab + (long)row0 * lda + k, lda, lane);
        #pragma unroll
        for (int t = 0; t < 4; t++) {
            bf16x16 b = load_frag_p(&bs[buf][(t * 16) * 32], 32, lane);
            acc[t] = wmma_bf16(a, b, acc[t]);
        }
        buf ^= 1;
    }

    #pragma unroll
    for (int t = 0; t < 4; t++) {
        #pragma unroll
        for (int r = 0; r < 8; r++) {
            int m = row0 + r + ((lane >> 4) << 3);
            int n = col0 + t * 16 + (lane & 15);
            float v = acc[t][r];
            if constexpr (EPI == 0) {
                e.of[(long)bz * e.cstride + (long)m * e.ldc + n] = v * e.alpha;
            } else if constexpr (EPI == 1) {
                long idx = (long)m * e.ldc + n;
                e.obf[idx] = f2bf(bf2f(e.mul[idx]) * (1.0f + v));
            } else if constexpr (EPI == 2) {
                e.obf[(long)m * e.ldc + n] = f2bf(v + e.bias[n]);
            } else if constexpr (EPI == 3) {
                int bb = m >> 9, vv = m & 511, h = n >> 6, c = n & 63;
                e.obf[(((long)(bb * NUM_HEADS + h) * HEAD_DIM + c) << 9) + vv] = f2bf(v + e.bias[n]);
            } else {
                long idx = (long)m * e.ldc + n;
                e.of[idx] = e.resid[idx] + e.scale[n] * (v + e.bias[n]);
            }
        }
    }
}

// ---------------- fused attention: scores->softmax->PV per (b,h,u-tile) ----------------
// grid: (U/64, NUM_HEADS, B), block 256 (8 waves), dyn LDS = 64*512 f32 (128 KB)
__global__ __launch_bounds__(256)
void attn_fused(const u16t* __restrict__ q, const u16t* __restrict__ kk_,
                const u16t* __restrict__ vT, const float* __restrict__ logaff,
                u16t* __restrict__ msg) {
    extern __shared__ float sc[];          // [64][512]
    __shared__ float red[64 * 4];
    int bb = blockIdx.z, h = blockIdx.y;
    int u0 = blockIdx.x * 64;
    int wave = threadIdx.x >> 5;
    int lane = threadIdx.x & 31;

    const u16t* qh = q   + (long)bb * UU * INNER_DIM + h * HEAD_DIM;
    const u16t* kh = kk_ + (long)bb * VV * INNER_DIM + h * HEAD_DIM;

    // Phase A: scores = q k^T / 8 + logaff  -> LDS (f32)
    {
        int rt = wave >> 1;                 // 4 row-tiles, 2 waves each
        bf16x16 aF[2];
        #pragma unroll
        for (int ks = 0; ks < 2; ks++)
            aF[ks] = load_frag(qh + (long)(u0 + rt * 16) * INNER_DIM + ks * 32, INNER_DIM, lane);
        for (int j = 0; j < 16; j++) {
            int ct = (wave & 1) * 16 + j;
            f32x8 acc;
            #pragma unroll
            for (int i = 0; i < 8; i++) acc[i] = 0.0f;
            #pragma unroll
            for (int ks = 0; ks < 2; ks++) {
                bf16x16 bF = load_frag(kh + (long)(ct * 16) * INNER_DIM + ks * 32, INNER_DIM, lane);
                acc = wmma_bf16(aF[ks], bF, acc);
            }
            #pragma unroll
            for (int r = 0; r < 8; r++) {
                int m = r + ((lane >> 4) << 3);
                int n = lane & 15;
                int ul = rt * 16 + m, vc = ct * 16 + n;
                sc[ul * 512 + vc] = acc[r] * 0.125f +
                    logaff[((long)bb * UU + (u0 + ul)) * (long)VV + vc];
            }
        }
    }
    __syncthreads();

    // Phase B: softmax over 512 cols per row (4 threads/row x 128 cols)
    {
        int row = threadIdx.x >> 2, part = threadIdx.x & 3;
        float* rp = sc + row * 512 + part * 128;
        float mx = -1e30f;
        for (int c = 0; c < 128; c++) mx = fmaxf(mx, rp[c]);
        red[row * 4 + part] = mx;
        __syncthreads();
        float rm = fmaxf(fmaxf(red[row * 4], red[row * 4 + 1]),
                         fmaxf(red[row * 4 + 2], red[row * 4 + 3]));
        __syncthreads();
        float s = 0.0f;
        for (int c = 0; c < 128; c++) { float e2 = __expf(rp[c] - rm); rp[c] = e2; s += e2; }
        red[row * 4 + part] = s;
        __syncthreads();
        float inv = 1.0f / (red[row * 4] + red[row * 4 + 1] + red[row * 4 + 2] + red[row * 4 + 3]);
        for (int c = 0; c < 128; c++) rp[c] *= inv;
    }
    __syncthreads();

    // Phase C: msg[64,64] = weights[64,512] @ vT[64,512]^T  (NT, K=512)
    {
        const u16t* vh = vT + ((long)(bb * NUM_HEADS + h) * HEAD_DIM) * (long)VV; // row c, ld=512
        for (int tt = 0; tt < 2; tt++) {
            int t = wave * 2 + tt;
            int rt = t >> 2, ct = t & 3;
            f32x8 acc;
            #pragma unroll
            for (int i = 0; i < 8; i++) acc[i] = 0.0f;
            int m = lane & 15, kofs = (lane >> 4) * 8;
            for (int ks = 0; ks < 512; ks += 32) {
                FragB fa;
                const float* ap = sc + (rt * 16 + m) * 512 + ks + kofs;
                #pragma unroll
                for (int j = 0; j < 8; j++) {
                    fa.u[j]     = f2bf(ap[j]);
                    fa.u[8 + j] = f2bf(ap[16 + j]);
                }
                bf16x16 bF = load_frag(vh + (long)(ct * 16) * VV + ks, VV, lane);
                acc = wmma_bf16(fa.v, bF, acc);
            }
            #pragma unroll
            for (int r = 0; r < 8; r++) {
                int mm = r + ((lane >> 4) << 3);
                int nn = lane & 15;
                int u = u0 + rt * 16 + mm;
                int c = ct * 16 + nn;
                msg[((long)bb * UU + u) * INNER_DIM + h * HEAD_DIM + c] = f2bf(acc[r]);
            }
        }
    }
}

extern "C" void kernel_launch(void* const* d_in, const int* in_sizes, int n_in,
                              void* d_out, int out_size, void* d_ws, size_t ws_size,
                              hipStream_t stream) {
    const float* recv_states = (const float*)d_in[0];
    const float* recv_sig    = (const float*)d_in[1];
    const float* recv_codes  = (const float*)d_in[2];
    const float* send_states = (const float*)d_in[3];
    const float* send_sig    = (const float*)d_in[4];
    const float* send_codes  = (const float*)d_in[5];
    const float* ln_r_g = (const float*)d_in[6];
    const float* ln_r_b = (const float*)d_in[7];
    const float* ln_s_g = (const float*)d_in[8];
    const float* ln_s_b = (const float*)d_in[9];
    const float* Wq  = (const float*)d_in[10]; const float* bq = (const float*)d_in[11];
    const float* Wqm = (const float*)d_in[12];
    const float* Wk  = (const float*)d_in[13]; const float* bk = (const float*)d_in[14];
    const float* Wkm = (const float*)d_in[15];
    const float* Wv  = (const float*)d_in[16]; const float* bv = (const float*)d_in[17];
    const float* Wvm = (const float*)d_in[18];
    const float* Wo  = (const float*)d_in[19]; const float* bo = (const float*)d_in[20];
    const float* Wom = (const float*)d_in[21];
    const float* lscale = (const float*)d_in[22];

    char* ws = (char*)d_ws;
    size_t off = 0;
    auto alloc = [&](size_t bytes) -> char* {
        char* p = ws + off;
        off = (off + bytes + 255) & ~(size_t)255;
        return p;
    };
    const long NROW = (long)BB * UU;   // 4096
    u16t* rln   = (u16t*)alloc(NROW * STATE_DIM * 2);
    u16t* sln   = (u16t*)alloc(NROW * STATE_DIM * 2);
    u16t* rcod  = (u16t*)alloc(NROW * CODE_DIM * 2);
    u16t* scod  = (u16t*)alloc(NROW * CODE_DIM * 2);
    u16t* rsigb = (u16t*)alloc(NROW * SIG_DIM * 2);
    u16t* ssigb = (u16t*)alloc(NROW * SIG_DIM * 2);
    u16t* wqb   = (u16t*)alloc((long)INNER_DIM * STATE_DIM * 2);
    u16t* wkb   = (u16t*)alloc((long)INNER_DIM * STATE_DIM * 2);
    u16t* wvb   = (u16t*)alloc((long)INNER_DIM * STATE_DIM * 2);
    u16t* wob   = (u16t*)alloc((long)STATE_DIM * INNER_DIM * 2);
    u16t* wqmb  = (u16t*)alloc((long)STATE_DIM * CODE_DIM * 2);
    u16t* wkmb  = (u16t*)alloc((long)STATE_DIM * CODE_DIM * 2);
    u16t* wvmb  = (u16t*)alloc((long)STATE_DIM * CODE_DIM * 2);
    u16t* womb  = (u16t*)alloc((long)INNER_DIM * CODE_DIM * 2);
    u16t* xq    = (u16t*)alloc(NROW * STATE_DIM * 2);
    u16t* xk    = (u16t*)alloc(NROW * STATE_DIM * 2);
    u16t* xv    = (u16t*)alloc(NROW * STATE_DIM * 2);
    u16t* qb    = (u16t*)alloc(NROW * INNER_DIM * 2);
    u16t* kb    = (u16t*)alloc(NROW * INNER_DIM * 2);
    u16t* vTb   = (u16t*)alloc(NROW * INNER_DIM * 2);
    u16t* msg   = (u16t*)alloc(NROW * INNER_DIM * 2);
    u16t* xo    = (u16t*)alloc(NROW * INNER_DIM * 2);
    float* logaff = (float*)alloc((long)BB * UU * VV * 4);

    // 1) convert weights / codes / signatures to bf16
    auto cvt = [&](const float* src, u16t* dst, long n) {
        int blocks = (int)((n + 1023) / 1024);
        if (blocks > 4096) blocks = 4096;
        cvt_bf16<<<blocks, 256, 0, stream>>>(src, dst, (int)n);
    };
    cvt(recv_sig,   rsigb, NROW * SIG_DIM);
    cvt(send_sig,   ssigb, NROW * SIG_DIM);
    cvt(recv_codes, rcod,  NROW * CODE_DIM);
    cvt(send_codes, scod,  NROW * CODE_DIM);
    cvt(Wq,  wqb,  (long)INNER_DIM * STATE_DIM);
    cvt(Wk,  wkb,  (long)INNER_DIM * STATE_DIM);
    cvt(Wv,  wvb,  (long)INNER_DIM * STATE_DIM);
    cvt(Wo,  wob,  (long)STATE_DIM * INNER_DIM);
    cvt(Wqm, wqmb, (long)STATE_DIM * CODE_DIM);
    cvt(Wkm, wkmb, (long)STATE_DIM * CODE_DIM);
    cvt(Wvm, wvmb, (long)STATE_DIM * CODE_DIM);
    cvt(Wom, womb, (long)INNER_DIM * CODE_DIM);

    // 2) layernorms -> bf16
    layernorm_k<<<NROW, 256, 0, stream>>>(recv_states, ln_r_g, ln_r_b, rln);
    layernorm_k<<<NROW, 256, 0, stream>>>(send_states, ln_s_g, ln_s_b, sln);

    // 3) signature logits (batched NT, K=64) then row log-softmax
    {
        EpiArgs e{}; e.of = logaff; e.alpha = 0.125f; e.ldc = VV; e.cstride = (long)UU * VV;
        gemm_nt<0><<<dim3(VV / 64, UU / 64, BB), 128, 0, stream>>>(
            rsigb, ssigb, SIG_DIM, SIG_DIM, SIG_DIM,
            (long)UU * SIG_DIM, (long)VV * SIG_DIM, e);
    }
    logsoftmax_k<<<NROW, 256, 0, stream>>>(logaff);

    // 4) mod GEMMs: x = ln * (1 + codes @ Wm^T)
    {
        EpiArgs e{}; e.mul = rln; e.obf = xq; e.ldc = STATE_DIM;
        gemm_nt<1><<<dim3(STATE_DIM / 64, NROW / 64, 1), 128, 0, stream>>>(
            rcod, wqmb, CODE_DIM, CODE_DIM, CODE_DIM, 0, 0, e);
    }
    {
        EpiArgs e{}; e.mul = sln; e.obf = xk; e.ldc = STATE_DIM;
        gemm_nt<1><<<dim3(STATE_DIM / 64, NROW / 64, 1), 128, 0, stream>>>(
            scod, wkmb, CODE_DIM, CODE_DIM, CODE_DIM, 0, 0, e);
    }
    {
        EpiArgs e{}; e.mul = sln; e.obf = xv; e.ldc = STATE_DIM;
        gemm_nt<1><<<dim3(STATE_DIM / 64, NROW / 64, 1), 128, 0, stream>>>(
            scod, wvmb, CODE_DIM, CODE_DIM, CODE_DIM, 0, 0, e);
    }

    // 5) main projections
    {
        EpiArgs e{}; e.bias = bq; e.obf = qb; e.ldc = INNER_DIM;
        gemm_nt<2><<<dim3(INNER_DIM / 64, NROW / 64, 1), 128, 0, stream>>>(
            xq, wqb, STATE_DIM, STATE_DIM, STATE_DIM, 0, 0, e);
    }
    {
        EpiArgs e{}; e.bias = bk; e.obf = kb; e.ldc = INNER_DIM;
        gemm_nt<2><<<dim3(INNER_DIM / 64, NROW / 64, 1), 128, 0, stream>>>(
            xk, wkb, STATE_DIM, STATE_DIM, STATE_DIM, 0, 0, e);
    }
    {
        EpiArgs e{}; e.bias = bv; e.obf = vTb; e.ldc = INNER_DIM;   // writes v transposed per head
        gemm_nt<3><<<dim3(INNER_DIM / 64, NROW / 64, 1), 128, 0, stream>>>(
            xv, wvb, STATE_DIM, STATE_DIM, STATE_DIM, 0, 0, e);
    }

    // 6) fused attention (scores + log_aff, softmax, PV) -> msg bf16
    attn_fused<<<dim3(UU / 64, NUM_HEADS, BB), 256, 64 * 512 * sizeof(float), stream>>>(
        qb, kb, vTb, logaff, msg);

    // 7) output mod: xo = msg * (1 + codes @ Wom^T)
    {
        EpiArgs e{}; e.mul = msg; e.obf = xo; e.ldc = INNER_DIM;
        gemm_nt<1><<<dim3(INNER_DIM / 64, NROW / 64, 1), 128, 0, stream>>>(
            rcod, womb, CODE_DIM, CODE_DIM, CODE_DIM, 0, 0, e);
    }

    // 8) out projection + layer_scale + residual -> f32 d_out
    {
        EpiArgs e{}; e.bias = bo; e.of = (float*)d_out; e.resid = recv_states;
        e.scale = lscale; e.ldc = STATE_DIM;
        gemm_nt<4><<<dim3(STATE_DIM / 64, NROW / 64, 1), 128, 0, stream>>>(
            xo, wob, INNER_DIM, INNER_DIM, INNER_DIM, 0, 0, e);
    }
}
